// CopyNetwork_4569845203517
// MI455X (gfx1250) — matrix operand order repaired and checked by reading, not compiled, not run on hardware
//
#include <hip/hip_runtime.h>
#include <hip/hip_bf16.h>
#include <math.h>

// ---------------------------------------------------------------------------
// CopyNetwork head for MI455X (gfx1250, wave32, WMMA)
//   out0 = concat(log_softmax(H @ W_gen + b_gen), log(clip(segsum(scores*gate))))
//   out1 = gate = sigmoid(raw @ W_copy + b_copy)
// ---------------------------------------------------------------------------

typedef __attribute__((ext_vector_type(16))) _Float16 v16h;
typedef __attribute__((ext_vector_type(8)))  _Float16 v8h;
typedef __attribute__((ext_vector_type(4)))  _Float16 v4h;
typedef __attribute__((ext_vector_type(8)))  float    v8f;

#define TLEN  128
#define BATCH 32
#define SRC   512
#define DDIM  512
#define TGT   32000
#define CNUM  1024
#define MROWS (TLEN * BATCH)   // 4096
#define LDC   (TGT + CNUM)     // 33024

#define BM  128
#define BN  128
#define BK  32
#define BKP 40                 // padded LDS row stride in halves (80B -> conflict-free b128)

// ---------------------------------------------------------------------------
// Kernel 1: logits GEMM, f32 -> f16 in LDS, f32 accumulate via WMMA.
// grid = (TGT/BN, MROWS/BM) = (250, 32); block = 256 threads = 8 waves.
// Each wave owns a 32(M) x 64(N) subtile = 2x4 fragments of 16x16.
// Software-pipelined: next K-block's global loads overlap the WMMA clause.
// ---------------------------------------------------------------------------
__global__ __launch_bounds__(256)
void gen_logits_gemm(const float* __restrict__ A,     // (MROWS, DDIM) row-major
                     const float* __restrict__ B,     // (DDIM, TGT)   row-major
                     const float* __restrict__ bias,  // (TGT)
                     float* __restrict__ out)         // (MROWS, LDC)
{
    __shared__ _Float16 As[BM * BKP];   // [m][k], k contiguous
    __shared__ _Float16 Bs[BN * BKP];   // [n][k], k contiguous (transposed stage)

    const int tid  = threadIdx.x;
    const int lane = tid & 31;
    const int w    = tid >> 5;          // wave 0..7
    const int wm   = w & 3;             // 4 M-subtiles of 32 rows
    const int wn   = w >> 2;            // 2 N-subtiles of 64 cols
    const int bn0  = blockIdx.x * BN;
    const int bm0  = blockIdx.y * BM;
    const int l16  = lane & 15;
    const int hsel = lane >> 4;         // lane half select

    // per-thread staging coordinates (fixed across K loop)
    size_t aIdx[4], bIdx[4];
    int    aLds[4], bK[4], bN[4];
    #pragma unroll
    for (int j = 0; j < 4; ++j) {
        const int g  = tid + 256 * j;            // 0..1023 float4 groups
        const int m  = g >> 3;                   // 0..127
        const int kq = (g & 7) << 2;             // 0,4,..,28
        aIdx[j] = (size_t)(bm0 + m) * DDIM + kq;
        aLds[j] = m * BKP + kq;
        const int k  = g >> 5;                   // 0..31
        const int nq = (g & 31) << 2;            // 0..124
        bIdx[j] = (size_t)k * TGT + bn0 + nq;
        bK[j] = k; bN[j] = nq;
    }

    v8f acc[2][4];
    #pragma unroll
    for (int i = 0; i < 2; ++i)
        #pragma unroll
        for (int j = 0; j < 4; ++j)
            acc[i][j] = (v8f){0.f,0.f,0.f,0.f,0.f,0.f,0.f,0.f};

    // prologue: prefetch K-block 0 into registers
    float4 aR[4], bR[4];
    #pragma unroll
    for (int j = 0; j < 4; ++j) {
        aR[j] = *(const float4*)(A + aIdx[j]);
        bR[j] = *(const float4*)(B + bIdx[j]);
    }

    for (int kb = 0; kb < DDIM; kb += BK) {
        // ---- commit staged registers to LDS (f32 -> f16)
        #pragma unroll
        for (int j = 0; j < 4; ++j) {
            const v4h pa = {(_Float16)aR[j].x, (_Float16)aR[j].y,
                            (_Float16)aR[j].z, (_Float16)aR[j].w};
            *(v4h*)&As[aLds[j]] = pa;                 // ds_store_b64 (8B aligned)
            Bs[(bN[j] + 0) * BKP + bK[j]] = (_Float16)bR[j].x;
            Bs[(bN[j] + 1) * BKP + bK[j]] = (_Float16)bR[j].y;
            Bs[(bN[j] + 2) * BKP + bK[j]] = (_Float16)bR[j].z;
            Bs[(bN[j] + 3) * BKP + bK[j]] = (_Float16)bR[j].w;
        }
        __syncthreads();

        // ---- prefetch next K-block while this one computes
        if (kb + BK < DDIM) {
            #pragma unroll
            for (int j = 0; j < 4; ++j) {
                aR[j] = *(const float4*)(A + aIdx[j] + (kb + BK));
                bR[j] = *(const float4*)(B + bIdx[j] + (size_t)(kb + BK) * TGT);
            }
        }

        // ---- A fragments: 16x32 f16. lanes 0-15 hold K 0..7 / 16..23,
        //      lanes 16-31 hold K 8..15 / 24..31 (ISA 7.12.2 layout).
        v16h afrag[2];
        #pragma unroll
        for (int i = 0; i < 2; ++i) {
            const int Mr = wm * 32 + i * 16 + l16;
            const v8h lo = *(const v8h*)&As[Mr * BKP + hsel * 8];
            const v8h hi = *(const v8h*)&As[Mr * BKP + 16 + hsel * 8];
            afrag[i] = __builtin_shufflevector(lo, hi,
                0,1,2,3,4,5,6,7,8,9,10,11,12,13,14,15);
        }
        // ---- B fragments: 32x16 f16. N = lane%16; lanes 0-15 K 0..7 / 16..23,
        //      lanes 16-31 K 8..15 / 24..31 (same K order as A).
        v16h bfrag[4];
        #pragma unroll
        for (int j = 0; j < 4; ++j) {
            const int Nc = wn * 64 + j * 16 + l16;
            const v8h lo = *(const v8h*)&Bs[Nc * BKP + hsel * 8];
            const v8h hi = *(const v8h*)&Bs[Nc * BKP + 16 + hsel * 8];
            bfrag[j] = __builtin_shufflevector(lo, hi,
                0,1,2,3,4,5,6,7,8,9,10,11,12,13,14,15);
        }

        #pragma unroll
        for (int i = 0; i < 2; ++i)
            #pragma unroll
            for (int j = 0; j < 4; ++j)
                acc[i][j] = __builtin_amdgcn_wmma_f32_16x16x32_f16(
                    false, afrag[i], false, bfrag[j],
                    (short)0, acc[i][j], false, false);

        __syncthreads();
    }

    // ---- epilogue: logits + bias into the gen slice of the output row
    #pragma unroll
    for (int i = 0; i < 2; ++i) {
        #pragma unroll
        for (int j = 0; j < 4; ++j) {
            const int gn = bn0 + wn * 64 + j * 16 + l16;
            const float bv = bias[gn];
            #pragma unroll
            for (int r = 0; r < 8; ++r) {
                const int gm = bm0 + wm * 32 + i * 16 + hsel * 8 + r;  // C/D: M = r + 8*hsel
                out[(size_t)gm * LDC + gn] = acc[i][j][r] + bv;
            }
        }
    }
}

// ---------------------------------------------------------------------------
// Kernel 2: in-place log_softmax over the first TGT entries of each row.
// grid = MROWS, block = 256.
// ---------------------------------------------------------------------------
__global__ __launch_bounds__(256)
void row_log_softmax(float* __restrict__ out)
{
    __shared__ float red[256];
    float* row = out + (size_t)blockIdx.x * LDC;
    const int tid = threadIdx.x;

    float m = -INFINITY;
    for (int i = tid; i < TGT; i += 256) m = fmaxf(m, row[i]);
    red[tid] = m; __syncthreads();
    for (int s = 128; s > 0; s >>= 1) {
        if (tid < s) red[tid] = fmaxf(red[tid], red[tid + s]);
        __syncthreads();
    }
    m = red[0];
    __syncthreads();

    float sum = 0.f;
    for (int i = tid; i < TGT; i += 256) sum += __expf(row[i] - m);
    red[tid] = sum; __syncthreads();
    for (int s = 128; s > 0; s >>= 1) {
        if (tid < s) red[tid] += red[tid + s];
        __syncthreads();
    }
    const float lse = m + __logf(red[0]);

    for (int i = tid; i < TGT; i += 256) row[i] -= lse;
}

// ---------------------------------------------------------------------------
// Kernel 3: copy gate + masked segment-sum into CNUM bins + log(clip).
// grid = MROWS (one block per (t,b)), block = 256.
// ---------------------------------------------------------------------------
__global__ __launch_bounds__(256)
void copy_path(const float* __restrict__ raw,     // (MROWS, DDIM)
               const float* __restrict__ scores,  // (MROWS, SRC)
               const float* __restrict__ Wc,      // (DDIM)
               const float* __restrict__ bc,      // (1)
               const int*   __restrict__ ids,     // (SRC, BATCH)
               const int*   __restrict__ pcnum,   // scalar
               const int*   __restrict__ punk,    // scalar
               float* __restrict__ out)
{
    __shared__ float bins[CNUM];
    __shared__ float red[256];
    const int r   = blockIdx.x;            // r = t*BATCH + b
    const int b   = r & (BATCH - 1);
    const int tid = threadIdx.x;
    const int unk  = *punk;
    const int cnum = *pcnum;

    // gate = sigmoid(raw[r,:] . Wc + bc)
    float p = 0.f;
    for (int i = tid; i < DDIM; i += 256)
        p += raw[(size_t)r * DDIM + i] * Wc[i];
    red[tid] = p; __syncthreads();
    for (int s = 128; s > 0; s >>= 1) {
        if (tid < s) red[tid] += red[tid + s];
        __syncthreads();
    }
    const float gate = 1.f / (1.f + __expf(-(red[0] + bc[0])));

    for (int c = tid; c < CNUM; c += 256) bins[c] = 0.f;
    __syncthreads();

    for (int s = tid; s < SRC; s += 256) {
        const int id = ids[s * BATCH + b];
        if (id != unk && id >= 0 && id < cnum)
            atomicAdd(&bins[id], scores[(size_t)r * SRC + s] * gate);  // ds_add_f32
    }
    __syncthreads();

    for (int c = tid; c < CNUM; c += 256) {
        float v = bins[c];
        v = fminf(fmaxf(v, 1e-6f), 1.f - 1e-6f);
        out[(size_t)r * LDC + TGT + c] = __logf(v);
    }
    if (tid == 0) out[(size_t)MROWS * LDC + r] = gate;  // copy_scores output
}

// ---------------------------------------------------------------------------
extern "C" void kernel_launch(void* const* d_in, const int* in_sizes, int n_in,
                              void* d_out, int out_size, void* d_ws, size_t ws_size,
                              hipStream_t stream)
{
    const float* raw    = (const float*)d_in[0];  // raw_decoder_hidden
    const float* hidden = (const float*)d_in[1];  // out_decoder_hidden
    /* d_in[2] = content (unused by the reference) */
    const float* scores = (const float*)d_in[3];
    const float* Wg     = (const float*)d_in[4];
    const float* bg     = (const float*)d_in[5];
    const float* Wc     = (const float*)d_in[6];
    const float* bc     = (const float*)d_in[7];
    const int*   ids    = (const int*)d_in[8];
    const int*   pcnum  = (const int*)d_in[9];
    const int*   punk   = (const int*)d_in[10];
    float* out = (float*)d_out;

    dim3 gGemm(TGT / BN, MROWS / BM);   // (250, 32)
    gen_logits_gemm<<<gGemm, 256, 0, stream>>>(hidden, Wg, bg, out);
    row_log_softmax<<<MROWS, 256, 0, stream>>>(out);
    copy_path<<<MROWS, 256, 0, stream>>>(raw, scores, Wc, bc, ids, pcnum, punk, out);
}
